// MultiHeadAttention_35905926595388
// MI455X (gfx1250) — matrix operand order repaired and checked by reading, not compiled, not run on hardware
//
#include <hip/hip_runtime.h>
#include <hip/hip_bf16.h>
#include <stdint.h>

#define B_  4
#define S_  1024
#define DIN 768
#define H_  12
#define DH  64

typedef __attribute__((ext_vector_type(16))) __bf16         v16bf;
typedef __attribute__((ext_vector_type(8)))  float          v8f;
typedef __attribute__((ext_vector_type(16))) unsigned short us16;
typedef __attribute__((ext_vector_type(4)))  unsigned int   u32x4;
typedef __attribute__((ext_vector_type(8)))  int            i32x8;
typedef __attribute__((ext_vector_type(4)))  int            i32x4;

#if __has_builtin(__builtin_amdgcn_tensor_load_to_lds) && \
    __has_builtin(__builtin_amdgcn_s_wait_tensorcnt)
#define USE_TDM 1
#else
#define USE_TDM 0
#endif

__device__ __forceinline__ unsigned short f2bf(float f) {
  unsigned int u = __builtin_bit_cast(unsigned int, f);
  u += 0x7FFFu + ((u >> 16) & 1u);            // round-to-nearest-even
  return (unsigned short)(u >> 16);
}
__device__ __forceinline__ v8f wmma_bf16(v16bf a, v16bf b, v8f c) {
  return __builtin_amdgcn_wmma_f32_16x16x32_bf16(false, a, false, b, (short)0, c,
                                                 false, false);
}
__device__ __forceinline__ v16bf ldfrag(const unsigned short* p) {
  return __builtin_bit_cast(v16bf, *(const us16*)p);
}
// A-fragment K permutation within a 32-element chunk:
// pos(k): {0..7}->0..7, {16..23}->8..15, {8..15}->16..23, {24..31}->24..31
__device__ __forceinline__ int p32(int k) {
  return (k & 7) | ((k & 16) >> 1) | ((k & 8) << 1);
}

#if USE_TDM
// 2D bf16 tile load 64x64 (row stride 64 elements) global -> LDS via TDM.
__device__ __forceinline__ void tdm_load_tile64(unsigned lds_byte_addr,
                                                const unsigned short* gptr) {
  unsigned long long ga = (unsigned long long)(uintptr_t)gptr;
  u32x4 g0 = {1u,                                  // count=1 valid descriptor
              lds_byte_addr,                       // D#.lds_addr
              (unsigned)ga,                        // global_addr[31:0]
              (unsigned)((ga >> 32) & 0x1FFFFFFu)  // global_addr[56:32]
                  | (2u << 30)};                   // type = 2 ("image")
  i32x8 g1;
  g1[0] = 1 << 16;          // workgroup_mask=0, data_size=1 (2 bytes)
  g1[1] = (int)(64u << 16); // tensor_dim0[15:0]=64 (atomic_barrier_addr=0)
  g1[2] = (int)(64u << 16); // tensor_dim0 hi=0, tensor_dim1[15:0]=64
  g1[3] = (int)(64u << 16); // tensor_dim1 hi=0, tile_dim0=64
  g1[4] = 64;               // tile_dim1=64, tile_dim2=0
  g1[5] = 64;               // tensor_dim0_stride lo = 64 elements
  g1[6] = 0;                // stride0 hi, tensor_dim1_stride lo (unused, 2D)
  g1[7] = 0;
  i32x4 z4 = {0, 0, 0, 0};
#if defined(__clang_major__) && (__clang_major__ >= 23)
  i32x8 z8 = {0, 0, 0, 0, 0, 0, 0, 0};
  __builtin_amdgcn_tensor_load_to_lds(g0, g1, z4, z4, z8, 0);
#else
  __builtin_amdgcn_tensor_load_to_lds(g0, g1, z4, z4, 0);
#endif
}
#endif

// ---------------------------------------------------------------------------
// Kernel 1: QKV projection into bf16 workspace.
//   Q stored [B,H,S,DH] row-major
//   K stored [B,H,S,DH] row-major
//   V stored [B,H,DH,S] transposed
// One block = one 64x64 tile of one (z, head). 8 waves, each 16x32 of C.
// ---------------------------------------------------------------------------
__global__ __launch_bounds__(256)
void qkv_proj_kernel(const float* __restrict__ x,
                     const float* __restrict__ Wq,
                     const float* __restrict__ Wk,
                     const float* __restrict__ Wv,
                     unsigned short* __restrict__ Qw,
                     unsigned short* __restrict__ Kw,
                     unsigned short* __restrict__ Vw) {
  const int h = blockIdx.y;
  const int z = blockIdx.z;
  const float* W = (z == 0) ? Wq : (z == 1) ? Wk : Wv;
  W += (size_t)h * DIN * DH;
  unsigned short* Out = (z == 0) ? Qw : (z == 1) ? Kw : Vw;

  const int rowBase = blockIdx.x * 64;       // row in flattened [B*S, DIN]
  const int tid  = threadIdx.x;
  const int wave = tid >> 5;
  const int lane = tid & 31;
  const int m    = lane & 15;
  const int hi   = lane >> 4;
  const int strip = wave >> 1;               // 0..3 (16-row strip)
  const int colh  = wave & 1;                // 0..1 (32-col half)

  __shared__ __align__(16) unsigned short Xs[64 * 32];   // permuted K-order
  __shared__ __align__(16) unsigned short Wt[64 * 32];   // transposed [n][k]

  v8f c0 = {}; v8f c1 = {};

  // staging coordinates (loop-invariant parts hoisted)
  const int xc = tid & 31;            // X col handled by this thread
  const int xr0 = tid >> 5;           // X row base (step 8)
  const int xp = p32(xc);
  const int wc = tid & 63;            // W col handled by this thread
  const int wr0 = tid >> 6;           // W row base (step 4)

  for (int kc = 0; kc < DIN / 32; ++kc) {
    const int kg = kc * 32;
    // Stage X tile (64x32) as bf16, K-permuted so A-frags are contiguous
#pragma unroll
    for (int i = 0; i < 8; ++i) {
      int r = xr0 + i * 8;
      Xs[r * 32 + xp] = f2bf(x[(size_t)(rowBase + r) * DIN + kg + xc]);
    }
    // Stage W tile (32x64) transposed -> Wt[n][k] so B-frags are contiguous
#pragma unroll
    for (int i = 0; i < 8; ++i) {
      int r = wr0 + i * 4;
      Wt[wc * 32 + p32(r)] = f2bf(W[(size_t)(kg + r) * DH + wc]);
    }
    __syncthreads();

    v16bf a  = ldfrag(&Xs[(strip * 16 + m) * 32 + hi * 16]);
    v16bf b0 = ldfrag(&Wt[(colh * 32 + m) * 32 + hi * 16]);
    v16bf b1 = ldfrag(&Wt[(colh * 32 + 16 + m) * 32 + hi * 16]);
    c0 = wmma_bf16(a, b0, c0);
    c1 = wmma_bf16(a, b1, c1);
    __syncthreads();
  }

  // Epilogue. C layout: row = r + 8*hi, col = lane&15 per 16-col subtile.
  // rowBase is a multiple of 64, so all 8 rows share the same batch bb.
  const int row0 = rowBase + strip * 16 + hi * 8;
  const int bb = row0 >> 10;
  const int s0 = row0 & 1023;
  if (z == 2) {                       // V: transposed [DH][S]
    size_t o  = (size_t)(bb * H_ + h) * DH * S_ + s0;
    size_t o0 = o + (size_t)(colh * 32 + m) * S_;
    size_t o1 = o + (size_t)(colh * 32 + 16 + m) * S_;
#pragma unroll
    for (int r = 0; r < 8; ++r) {
      Out[o0 + r] = f2bf(c0[r]);
      Out[o1 + r] = f2bf(c1[r]);
    }
  } else if (z == 0) {                // Q: row-major
    size_t o = ((size_t)(bb * H_ + h) * S_ + s0) * DH + colh * 32;
    const int q0 = m, q1 = 16 + m;
#pragma unroll
    for (int r = 0; r < 8; ++r) {
      Out[o + (size_t)r * DH + q0] = f2bf(c0[r]);
      Out[o + (size_t)r * DH + q1] = f2bf(c1[r]);
    }
  } else {                            // K: row-major
    size_t o = ((size_t)(bb * H_ + h) * S_ + s0) * DH + colh * 32 + m;
#pragma unroll
    for (int r = 0; r < 8; ++r) {
      Out[o + (size_t)r * DH]      = f2bf(c0[r]);
      Out[o + (size_t)r * DH + 16] = f2bf(c1[r]);
    }
  }
}

// ---------------------------------------------------------------------------
// Kernel 2: causal flash attention. One block = one (b, h, 64-query tile).
// 4 waves; wave w owns query rows w*16..w*16+15. K tiles arrive in LDS via
// TDM (double-buffered prefetch); V read straight from global (transposed),
// with global_prefetch of the next tile's rows.
// ---------------------------------------------------------------------------
__global__ __launch_bounds__(128)
void attn_kernel(const unsigned short* __restrict__ Qw,
                 const unsigned short* __restrict__ Kw,
                 const unsigned short* __restrict__ Vt,
                 float* __restrict__ out) {
  const int b  = blockIdx.z;
  const int h  = blockIdx.y;
  const int qt = blockIdx.x;
  const int qbase = qt * 64;
  const int tid  = threadIdx.x;
  const int wave = tid >> 5;
  const int lane = tid & 31;
  const int m    = lane & 15;
  const int hi   = lane >> 4;

  const size_t baseQK = (size_t)(b * H_ + h) * S_ * DH;  // [S][DH] planes
  const size_t baseV  = (size_t)(b * H_ + h) * DH * S_;  // [DH][S] plane

  __shared__ __align__(16) unsigned short Ks[2][64 * 64];
  __shared__ __align__(16) unsigned short Ps[4][16 * 64];

  // Q A-fragments: single 32B load each (Qw is row-major, contiguous k order)
  const int qrow = qbase + wave * 16 + m;
  v16bf qa[2];
#pragma unroll
  for (int ch = 0; ch < 2; ++ch)
    qa[ch] = ldfrag(Qw + baseQK + (size_t)qrow * DH + ch * 32 + hi * 16);

  float Mr[8], Lr[8];
#pragma unroll
  for (int r = 0; r < 8; ++r) { Mr[r] = -3.0e38f; Lr[r] = 0.0f; }
  v8f of[4] = {v8f{}, v8f{}, v8f{}, v8f{}};

  const float scale = 0.125f;                 // 1/sqrt(64)
  const int qg = qbase + wave * 16 + hi * 8;  // + r -> global query row

#if USE_TDM
  const unsigned ldsK0 = (unsigned)(uintptr_t)&Ks[0][0];
  const unsigned ldsK1 = (unsigned)(uintptr_t)&Ks[1][0];
  if (wave == 0) tdm_load_tile64(ldsK0, Kw + baseQK);   // prefetch tile 0
#endif

  for (int jt = 0; jt <= qt; ++jt) {
    const int kb = jt * 64;
#if USE_TDM
    if (wave == 0) {
      if (jt < qt) {   // prefetch next tile into the other buffer
        tdm_load_tile64((jt & 1) ? ldsK0 : ldsK1,
                        Kw + baseQK + (size_t)(kb + 64) * DH);
        __builtin_amdgcn_s_wait_tensorcnt(1);  // tile jt complete
      } else {
        __builtin_amdgcn_s_wait_tensorcnt(0);
      }
    }
    __syncthreads();
    const unsigned short* Kt = Ks[jt & 1];
#else
    {
      const uint4* src = (const uint4*)(Kw + baseQK + (size_t)kb * DH);
      uint4* dst = (uint4*)&Ks[0][0];
#pragma unroll
      for (int i = 0; i < 4; ++i) dst[i * 128 + tid] = src[i * 128 + tid];
    }
    __syncthreads();
    const unsigned short* Kt = Ks[0];
#endif

    // Prefetch next tile's V rows into cache while we compute this tile
    if (jt < qt) {
#pragma unroll
      for (int f = 0; f < 4; ++f)
        __builtin_prefetch(Vt + baseV + (size_t)(f * 16 + m) * S_ +
                           kb + 64 + hi * 32, 0, 3);
    }

    // S = Q @ K^T : B-frag = 32B contiguous LDS load
    v8f sf[4] = {v8f{}, v8f{}, v8f{}, v8f{}};
#pragma unroll
    for (int f = 0; f < 4; ++f) {
#pragma unroll
      for (int ch = 0; ch < 2; ++ch) {
        v16bf bk = ldfrag(Kt + (f * 16 + m) * 64 + ch * 32 + hi * 16);
        sf[f] = wmma_bf16(qa[ch], bk, sf[f]);
      }
    }

    const bool diag = (jt == qt);
    const int ppa = m;             // contiguous P-store positions
    const int ppb = 16 + m;
#pragma unroll
    for (int r = 0; r < 8; ++r) {
      const int qrg = qg + r;
      float v0 = sf[0][r] * scale;
      float v1 = sf[1][r] * scale;
      float v2 = sf[2][r] * scale;
      float v3 = sf[3][r] * scale;
      if (diag) {
        if (kb +  0 + m > qrg) v0 = -3.0e38f;
        if (kb + 16 + m > qrg) v1 = -3.0e38f;
        if (kb + 32 + m > qrg) v2 = -3.0e38f;
        if (kb + 48 + m > qrg) v3 = -3.0e38f;
      }
      float tmax = fmaxf(fmaxf(v0, v1), fmaxf(v2, v3));
#pragma unroll
      for (int d = 1; d < 16; d <<= 1)
        tmax = fmaxf(tmax, __shfl_xor(tmax, d, 32));
      float mnew = fmaxf(Mr[r], tmax);
      float corr = __expf(Mr[r] - mnew);
      float p0 = __expf(v0 - mnew);
      float p1 = __expf(v1 - mnew);
      float p2 = __expf(v2 - mnew);
      float p3 = __expf(v3 - mnew);
      float rsum = (p0 + p1) + (p2 + p3);
#pragma unroll
      for (int d = 1; d < 16; d <<= 1)
        rsum += __shfl_xor(rsum, d, 32);
      Lr[r] = Lr[r] * corr + rsum;
      Mr[r] = mnew;
      const int prow = hi * 8 + r;
      Ps[wave][prow * 64 + ppa]      = f2bf(p0);
      Ps[wave][prow * 64 + ppb]      = f2bf(p1);
      Ps[wave][prow * 64 + 32 + ppa] = f2bf(p2);
      Ps[wave][prow * 64 + 32 + ppb] = f2bf(p3);
      of[0][r] *= corr; of[1][r] *= corr; of[2][r] *= corr; of[3][r] *= corr;
    }

    // O += P @ V : P A-frag from per-wave LDS (contiguous -> one 32B load),
    // V B-frag straight from global (transposed layout -> contiguous 32B)
#pragma unroll
    for (int ch = 0; ch < 2; ++ch) {
      v16bf pfrag = ldfrag(&Ps[wave][m * 64 + ch * 32 + hi * 16]);
#pragma unroll
      for (int f = 0; f < 4; ++f) {
        v16bf bv = ldfrag(Vt + baseV + (size_t)(f * 16 + m) * S_ +
                          kb + ch * 32 + hi * 16);
        of[f] = wmma_bf16(pfrag, bv, of[f]);
      }
    }
    __syncthreads();
  }

  // Epilogue: out[b, s, h*64 + col] in fp32 (base hoisted, r via offsets)
  float* op0 = out + ((size_t)b * S_ + qg) * (H_ * DH) + h * DH + m;
#pragma unroll
  for (int r = 0; r < 8; ++r) {
    const float inv = 1.0f / Lr[r];
    float* op = op0 + (size_t)r * (H_ * DH);
    op[ 0] = of[0][r] * inv;
    op[16] = of[1][r] * inv;
    op[32] = of[2][r] * inv;
    op[48] = of[3][r] * inv;
  }
}

// ---------------------------------------------------------------------------
extern "C" void kernel_launch(void* const* d_in, const int* in_sizes, int n_in,
                              void* d_out, int out_size, void* d_ws, size_t ws_size,
                              hipStream_t stream) {
  const float* x  = (const float*)d_in[0];
  const float* Wq = (const float*)d_in[1];
  const float* Wk = (const float*)d_in[2];
  const float* Wv = (const float*)d_in[3];
  float* out = (float*)d_out;

  const size_t plane = (size_t)B_ * H_ * S_ * DH;   // elems per bf16 tensor
  unsigned short* Qw = (unsigned short*)d_ws;
  unsigned short* Kw = Qw + plane;
  unsigned short* Vw = Kw + plane;

  dim3 gp(B_ * S_ / 64, H_, 3);
  qkv_proj_kernel<<<gp, 256, 0, stream>>>(x, Wq, Wk, Wv, Qw, Kw, Vw);

  dim3 ga(S_ / 64, H_, B_);
  attn_kernel<<<ga, 128, 0, stream>>>(Qw, Kw, Vw, out);
}
